// dct_channel_block_3513283248810
// MI455X (gfx1250) — matrix operand order repaired and checked by reading, not compiled, not run on hardware
//
#include <hip/hip_runtime.h>

typedef __attribute__((ext_vector_type(16))) _Float16 v16h;
typedef __attribute__((ext_vector_type(8)))  _Float16 v8h;
typedef __attribute__((ext_vector_type(8)))  float    v8f;

#define PI_F 3.14159265358979323846f
// Problem sizes: B=8192, C=96, L=96 -> ROWS = 786432 rows of length 96.
// DCT folded into FC1: Weff = Mdct @ W1 (96x192). FC2: 192 -> 96.
#define ROWS        786432
#define LQ          96
#define FQ          192
#define ROWS_PER_WG 128                   // 8 waves * 16 rows
#define NUM_WG      (ROWS / ROWS_PER_WG)  // 6144
#define W1_FRAGS    (3 * 12)              // Kchunks x Ntiles for FC1
#define W2_FRAGS    (6 * 6)               // Kchunks x Ntiles for FC2
#define WF_HALVES   (W1_FRAGS * 512)      // 18432 halves per weight array

// LDS layout (dynamic, 128000 B total):
//   [0, 36864)        FC1 B-fragments (f16)
//   [36864, 73728)    FC2 B-fragments (f16)
//   [73728, +8*6656)  per-wave staging slice (x_f16 / h_f16 / lr_f32 union)
//   [126976, +1024)   per-row layernorm stats
#define WEIGHT_BYTES (2 * WF_HALVES * 2)  // 73728
#define XS_STRIDE 104   // halves  (16 rows; padded so b128 gathers are conflict-free)
#define HS_STRIDE 200   // halves
#define LS_STRIDE 104   // floats
#define WAVE_BUF  6656  // bytes = max(16*104*2, 16*200*2, 16*104*4)
#define LDS_TOTAL (WEIGHT_BYTES + 8 * WAVE_BUF + 8 * 16 * 2 * 4)  // 128000

// ---------------------------------------------------------------------------
// Weight fragment storage order (both prep kernels write it, main kernel and
// the LDS copy consume it):  half_index = ((fi*2 + chunk)*32 + lane)*8 + e7
// where fragment fi covers B[kk,nn] = W[32c + kk, 16t + nn],
//   kk = 8*(lane/16) + 16*chunk + e7,  nn = lane%16.
// Per-lane lo/hi chunks are 16B with a 16B lane stride -> conflict-free
// ds_load_b128 and fully coalesced global accesses.
// ---------------------------------------------------------------------------

// Prep 1: Weff[n,f] = sum_k 2*cos(pi*(2n+1)k/192) * W1[k,f]  -> f16 fragments
__global__ __launch_bounds__(256) void prep_w1_kernel(const float* __restrict__ W1,
                                                      _Float16* __restrict__ wf) {
  int s = blockIdx.x * 256 + threadIdx.x;
  if (s >= WF_HALVES) return;
  int e7 = s & 7;
  int l  = (s >> 3) & 31;
  int k2 = (s >> 8) & 1;
  int fi = s >> 9;           // 0..35
  int e  = k2 * 8 + e7;
  int c  = fi / 12;
  int t  = fi % 12;
  int kk = ((l >> 4) << 3) + (k2 << 4) + e7;
  int nn = l & 15;
  int n  = c * 32 + kk;      // x-sample index (K dim of FC1), 0..95
  int f  = t * 16 + nn;      // hidden index, 0..191
  float acc = 0.0f;
  for (int k = 0; k < 96; ++k) {
    float ang = PI_F * (float)((2 * n + 1) * k) * (1.0f / 192.0f);
    acc += 2.0f * cosf(ang) * W1[k * FQ + f];
  }
  wf[s] = (_Float16)acc;
}

// Prep 2: W2 (192x96 f32) -> f16 fragments
__global__ __launch_bounds__(256) void prep_w2_kernel(const float* __restrict__ W2,
                                                      _Float16* __restrict__ w2f) {
  int s = blockIdx.x * 256 + threadIdx.x;
  if (s >= WF_HALVES) return;
  int e7 = s & 7;
  int l  = (s >> 3) & 31;
  int k2 = (s >> 8) & 1;
  int fi = s >> 9;           // 0..35
  int e  = k2 * 8 + e7;
  int c  = fi / 6;
  int t  = fi % 6;
  int kk = ((l >> 4) << 3) + (k2 << 4) + e7;
  int nn = l & 15;
  w2f[s] = (_Float16)W2[(c * 32 + kk) * LQ + t * 16 + nn];
}

// ---------------------------------------------------------------------------
// Fused main kernel: per-wave 16-row tile, weights staged once per block in LDS.
//   h  = relu(x_f16 @ Weff_f16)   (K=96  -> 3 chunks, 12 N-tiles, 36 WMMAs)
//   lr = sigmoid(h_f16 @ W2_f16)  (K=192 -> 6 chunks,  6 N-tiles, 36 WMMAs)
//   lr = layernorm(lr) ; out = x * lr   (f32)
// ---------------------------------------------------------------------------
__global__ __launch_bounds__(256) void fused_dct_ffn_kernel(
    const float* __restrict__ x, const _Float16* __restrict__ wg,
    const float* __restrict__ gamma, const float* __restrict__ beta,
    float* __restrict__ out) {
  extern __shared__ __align__(16) unsigned char dynsmem[];
  _Float16* wfl  = (_Float16*)dynsmem;              // FC1 fragments
  _Float16* w2fl = wfl + WF_HALVES;                 // FC2 fragments
  const int tid  = threadIdx.x;
  const int wave = tid >> 5;
  const int lane = tid & 31;
  unsigned char* mybuf = dynsmem + WEIGHT_BYTES + wave * WAVE_BUF;
  _Float16* xs = (_Float16*)mybuf;                  // 16 x XS_STRIDE halves
  _Float16* hs = (_Float16*)mybuf;                  // 16 x HS_STRIDE halves (reuse)
  float*    ls = (float*)mybuf;                     // 16 x LS_STRIDE floats (reuse)
  float* stats = (float*)(dynsmem + WEIGHT_BYTES + 8 * WAVE_BUF);  // [8][16][2]

  const long rowBase = (long)blockIdx.x * ROWS_PER_WG + wave * 16;
  const float* xg = x + rowBase * LQ;
  float* og = out + rowBase * LQ;

  // ---- Stage both weight-fragment arrays into LDS (73728 B, coalesced) ----
  {
    const uint4* src = (const uint4*)wg;
    uint4* dst = (uint4*)dynsmem;
#pragma unroll
    for (int i = 0; i < 18; ++i) {
      int idx = i * 256 + tid;
      dst[idx] = src[idx];
    }
  }

  // ---- Phase A: x tile (16x96 f32 = 384 float4, coalesced) -> f16 in LDS ----
#pragma unroll
  for (int i = 0; i < 12; ++i) {
    int idx = i * 32 + lane;
    int row = idx / 24;            // 24 float4 per 96-elem row
    int col = (idx % 24) * 4;
    float4 v = ((const float4*)xg)[idx];
    _Float16* d = xs + row * XS_STRIDE + col;
    d[0] = (_Float16)v.x; d[1] = (_Float16)v.y;
    d[2] = (_Float16)v.z; d[3] = (_Float16)v.w;
  }
  __syncthreads();

  // ---- A fragments for FC1 (16-bit A 16x32 layout) ----
  // lane holds row m=lane%16; K-halves [8h..8h+7] -> elems 0..7, [16+8h..] -> 8..15
  const int m  = lane & 15;
  const int hh = lane >> 4;
  v16h a[3];
#pragma unroll
  for (int c = 0; c < 3; ++c) {
    const _Float16* base = xs + m * XS_STRIDE + c * 32 + hh * 8;
    v8h lo = *(const v8h*)(base);
    v8h hi = *(const v8h*)(base + 16);
    a[c] = __builtin_shufflevector(lo, hi, 0, 1, 2, 3, 4, 5, 6, 7,
                                   8, 9, 10, 11, 12, 13, 14, 15);
  }
  __syncthreads();

  // ---- FC1: 3 K-chunks x 12 N-tiles = 36 WMMAs, B from LDS ----
  v8f zero = {0.f, 0.f, 0.f, 0.f, 0.f, 0.f, 0.f, 0.f};
  v8f acc[12];
#pragma unroll
  for (int t = 0; t < 12; ++t) acc[t] = zero;
#pragma unroll
  for (int c = 0; c < 3; ++c) {
#pragma unroll
    for (int t = 0; t < 12; ++t) {
      const _Float16* bb = wfl + (c * 12 + t) * 512 + lane * 8;
      v8h lo = *(const v8h*)(bb);
      v8h hi = *(const v8h*)(bb + 256);
      v16h b = __builtin_shufflevector(lo, hi, 0, 1, 2, 3, 4, 5, 6, 7,
                                       8, 9, 10, 11, 12, 13, 14, 15);
      acc[t] = __builtin_amdgcn_wmma_f32_16x16x32_f16(
          false, a[c], false, b, (short)0, acc[t], false, false);
    }
  }

  // ---- relu, C/D layout (col n=16t+m, row v+8*hh) -> row-major f16 h ----
#pragma unroll
  for (int t = 0; t < 12; ++t) {
    int n = t * 16 + m;
#pragma unroll
    for (int v = 0; v < 8; ++v) {
      float hv = acc[t][v];
      hv = hv > 0.0f ? hv : 0.0f;
      hs[(v + 8 * hh) * HS_STRIDE + n] = (_Float16)hv;
    }
  }
  __syncthreads();

  // ---- A fragments for FC2 (K=192 -> 6 chunks) ----
  v16h ha[6];
#pragma unroll
  for (int c = 0; c < 6; ++c) {
    const _Float16* base = hs + m * HS_STRIDE + c * 32 + hh * 8;
    v8h lo = *(const v8h*)(base);
    v8h hi = *(const v8h*)(base + 16);
    ha[c] = __builtin_shufflevector(lo, hi, 0, 1, 2, 3, 4, 5, 6, 7,
                                    8, 9, 10, 11, 12, 13, 14, 15);
  }

  // ---- FC2: 6 K-chunks x 6 N-tiles = 36 WMMAs, B from LDS ----
  v8f acc2[6];
#pragma unroll
  for (int t = 0; t < 6; ++t) acc2[t] = zero;
#pragma unroll
  for (int c = 0; c < 6; ++c) {
#pragma unroll
    for (int t = 0; t < 6; ++t) {
      const _Float16* bb = w2fl + (c * 6 + t) * 512 + lane * 8;
      v8h lo = *(const v8h*)(bb);
      v8h hi = *(const v8h*)(bb + 256);
      v16h b = __builtin_shufflevector(lo, hi, 0, 1, 2, 3, 4, 5, 6, 7,
                                       8, 9, 10, 11, 12, 13, 14, 15);
      acc2[t] = __builtin_amdgcn_wmma_f32_16x16x32_f16(
          false, ha[c], false, b, (short)0, acc2[t], false, false);
    }
  }
  __syncthreads();

  // ---- sigmoid -> f32 lr tile in LDS ----
#pragma unroll
  for (int t = 0; t < 6; ++t) {
    int n = t * 16 + m;
#pragma unroll
    for (int v = 0; v < 8; ++v) {
      float s = 1.0f / (1.0f + __expf(-acc2[t][v]));
      ls[(v + 8 * hh) * LS_STRIDE + n] = s;
    }
  }
  __syncthreads();

  // ---- layernorm stats: 2 lanes per row, 48 elems each, combine via shfl ----
  {
    int r  = lane >> 1;
    int h2 = lane & 1;
    const float4* lr4 = (const float4*)(ls + r * LS_STRIDE + h2 * 48);
    float s = 0.0f, sq = 0.0f;
#pragma unroll
    for (int i = 0; i < 12; ++i) {
      float4 v = lr4[i];
      s  += v.x + v.y + v.z + v.w;
      sq += v.x * v.x + v.y * v.y + v.z * v.z + v.w * v.w;
    }
    s  += __shfl_xor(s, 1);
    sq += __shfl_xor(sq, 1);
    float mu   = s * (1.0f / 96.0f);
    float var  = sq * (1.0f / 96.0f) - mu * mu;
    float rstd = rsqrtf(var + 1e-6f);
    if (h2 == 0) {
      stats[(wave * 16 + r) * 2 + 0] = mu;
      stats[(wave * 16 + r) * 2 + 1] = rstd;
    }
  }
  __syncthreads();

  // ---- out = x * ((lr - mu)*rstd*gamma + beta), coalesced float4 ----
#pragma unroll
  for (int i = 0; i < 12; ++i) {
    int idx  = i * 32 + lane;
    int row  = idx / 24;
    int col4 = idx % 24;
    float4 xv = ((const float4*)xg)[idx];
    float4 lv = *(const float4*)(ls + row * LS_STRIDE + col4 * 4);
    float4 g  = ((const float4*)gamma)[col4];
    float4 bb = ((const float4*)beta)[col4];
    float mu  = stats[(wave * 16 + row) * 2 + 0];
    float rs  = stats[(wave * 16 + row) * 2 + 1];
    float4 o;
    o.x = xv.x * ((lv.x - mu) * rs * g.x + bb.x);
    o.y = xv.y * ((lv.y - mu) * rs * g.y + bb.y);
    o.z = xv.z * ((lv.z - mu) * rs * g.z + bb.z);
    o.w = xv.w * ((lv.w - mu) * rs * g.w + bb.w);
    ((float4*)og)[idx] = o;
  }
}

extern "C" void kernel_launch(void* const* d_in, const int* in_sizes, int n_in,
                              void* d_out, int out_size, void* d_ws, size_t ws_size,
                              hipStream_t stream) {
  const float* x     = (const float*)d_in[0];
  const float* W1    = (const float*)d_in[1];
  const float* W2    = (const float*)d_in[2];
  const float* gamma = (const float*)d_in[3];
  const float* beta  = (const float*)d_in[4];
  float* out = (float*)d_out;

  _Float16* wf  = (_Float16*)d_ws;   // FC1 frags (36864 B) then FC2 frags (36864 B)
  _Float16* w2f = wf + WF_HALVES;

  prep_w1_kernel<<<72, 256, 0, stream>>>(W1, wf);
  prep_w2_kernel<<<72, 256, 0, stream>>>(W2, w2f);
  fused_dct_ffn_kernel<<<NUM_WG, 256, LDS_TOTAL, stream>>>(x, wf, gamma, beta, out);
}
